// Transformer_84207128805995
// MI455X (gfx1250) — matrix operand order repaired and checked by reading, not compiled, not run on hardware
//
#include <hip/hip_runtime.h>

// ---------- types ----------
typedef __bf16 bf16_t;
typedef __attribute__((ext_vector_type(16))) __bf16 v16bf;
typedef __attribute__((ext_vector_type(8)))  __bf16 v8bf;
typedef __attribute__((ext_vector_type(8)))  float  v8f;
typedef __attribute__((ext_vector_type(4)))  float  v4f;

#define WMMA_BF16(A_, B_, C_) \
  __builtin_amdgcn_wmma_f32_16x16x32_bf16(false, (A_), false, (B_), (short)0, (C_), false, false)

// ---------- problem constants ----------
#define BB    512
#define SS    200
#define EE    128
#define HH    4
#define DD    32
#define FFD   512
#define BHNUM (BB * HH)      // 2048
#define MROWS (BB * SS)      // 102400
#define SKEYS 224            // 7 chunks of 32 keys (>= 200)
#define VLD   256            // padded key stride for V^T layout (32B-aligned rows)

static __device__ inline v8f zero8() {
  v8f z;
#pragma unroll
  for (int i = 0; i < 8; ++i) z[i] = 0.0f;
  return z;
}

// A operand 16x32 bf16 from fp32 row-major source (on-the-fly convert).
// Lane's row pointer p; sel = lane>>4 selects K halves (contiguous order):
// elements 0..15 -> K = sel*16 + {0..15}.
static __device__ inline v16bf a_from_f32(const float* p, int sel) {
  const float* q = p + sel * 16;
  v4f x0 = *(const v4f*)(q + 0);
  v4f x1 = *(const v4f*)(q + 4);
  v4f x2 = *(const v4f*)(q + 8);
  v4f x3 = *(const v4f*)(q + 12);
  v16bf a;
#pragma unroll
  for (int i = 0; i < 4; ++i) {
    a[i]      = (bf16_t)x0[i];
    a[4 + i]  = (bf16_t)x1[i];
    a[8 + i]  = (bf16_t)x2[i];
    a[12 + i] = (bf16_t)x3[i];
  }
  return a;
}

// Same, from a bf16 row-major source (global or LDS).
static __device__ inline v16bf a_from_bf16(const bf16_t* p, int sel) {
  const bf16_t* q = p + sel * 16;
  v8bf lo = *(const v8bf*)(q);
  v8bf hi = *(const v8bf*)(q + 8);
  v16bf a;
#pragma unroll
  for (int i = 0; i < 8; ++i) { a[i] = lo[i]; a[8 + i] = hi[i]; }
  return a;
}

// row-wise reductions over the 16 lanes of a half-wave (C-layout rows)
static __device__ inline float red_max16(float v) {
  v = fmaxf(v, __shfl_xor(v, 1, 32));
  v = fmaxf(v, __shfl_xor(v, 2, 32));
  v = fmaxf(v, __shfl_xor(v, 4, 32));
  v = fmaxf(v, __shfl_xor(v, 8, 32));
  return v;
}
static __device__ inline float red_sum16(float v) {
  v += __shfl_xor(v, 1, 32);
  v += __shfl_xor(v, 2, 32);
  v += __shfl_xor(v, 4, 32);
  v += __shfl_xor(v, 8, 32);
  return v;
}

// ---------- kernel 0: fp32 -> bf16 weight cast ----------
__global__ void cvt_f32_bf16(const float* __restrict__ src,
                             unsigned short* __restrict__ dst, int n) {
  int i = blockIdx.x * 256 + threadIdx.x;
  if (i < n) dst[i] = __builtin_bit_cast(unsigned short, (bf16_t)src[i]);
}

// ---------- kernel 1: fused QKV projection ----------
// grid.x tiles M (64 rows / WG, 16 rows / wave), grid.y = {0:Q, 1:K, 2:V}
__global__ __launch_bounds__(128) void qkv_proj_kernel(
    const float* __restrict__ queries, const float* __restrict__ keys,
    const unsigned short* __restrict__ Wq_, const unsigned short* __restrict__ Wk_,
    const unsigned short* __restrict__ Wv_,
    const float* __restrict__ bq, const float* __restrict__ bk, const float* __restrict__ bv,
    unsigned short* __restrict__ Qh_, unsigned short* __restrict__ Kh_,
    unsigned short* __restrict__ Vt_) {
  const int z = blockIdx.y;
  const float* X = (z == 0) ? queries : keys;
  const bf16_t* W = (const bf16_t*)((z == 0) ? Wq_ : (z == 1) ? Wk_ : Wv_);
  const float* bias = (z == 0) ? bq : (z == 1) ? bk : bv;
  bf16_t* Qh = (bf16_t*)Qh_;
  bf16_t* Kh = (bf16_t*)Kh_;
  bf16_t* Vt = (bf16_t*)Vt_;

  const int wave = threadIdx.x >> 5, lane = threadIdx.x & 31;
  const int sel = lane >> 4, ln = lane & 15, rb = sel * 8;
  const int m0 = blockIdx.x * 64 + wave * 16;
  const float* Arow = X + (size_t)(m0 + ln) * EE;

  v8f acc[8];
#pragma unroll
  for (int t = 0; t < 8; ++t) acc[t] = zero8();

#pragma unroll
  for (int ks = 0; ks < 4; ++ks) {
    v16bf A = a_from_f32(Arow + ks * 32, sel);
#pragma unroll
    for (int t = 0; t < 8; ++t) {
      const bf16_t* Bp = W + (size_t)(t * 16 + ln) * EE + ks * 32 + sel * 16;
      acc[t] = WMMA_BF16(A, *(const v16bf*)Bp, acc[t]);
    }
  }

#pragma unroll
  for (int t = 0; t < 8; ++t) {
    const int n = t * 16 + ln;
    const float bn = bias[n];
    const int h = n >> 5, d = n & 31;
#pragma unroll
    for (int r = 0; r < 8; ++r) {
      const int m = m0 + rb + r;
      const int b = m / SS, s = m - b * SS;
      bf16_t v = (bf16_t)(acc[t][r] + bn);
      if (z == 2)      Vt[(((size_t)b * HH + h) * DD + d) * VLD + s] = v;      // V transposed
      else if (z == 0) Qh[(((size_t)b * HH + h) * SS + s) * DD + d] = v;
      else             Kh[(((size_t)b * HH + h) * SS + s) * DD + d] = v;
    }
  }
}

// ---------- kernel 2: flash attention per (b,h), residual fused ----------
__global__ __launch_bounds__(256) void attn_kernel(
    const unsigned short* __restrict__ Qh_, const unsigned short* __restrict__ Kh_,
    const unsigned short* __restrict__ Vt_, const unsigned char* __restrict__ kmask,
    const float* __restrict__ queries, float* __restrict__ outf) {
  __shared__ bf16_t sK[SKEYS * DD];     // [key][d], zero-padded beyond S
  __shared__ bf16_t sV[DD * SKEYS];     // [d][key], zero-padded beyond S
  __shared__ bf16_t sP[8][16 * 32];     // per-wave P re-layout buffer

  const bf16_t* Qh = (const bf16_t*)Qh_;
  const bf16_t* Kh = (const bf16_t*)Kh_;
  const bf16_t* Vt = (const bf16_t*)Vt_;

  const int bh = blockIdx.x;
  const int b = bh >> 2, h = bh & 3;
  const int tid = threadIdx.x;

  { // cooperative stage of K tile (uint = 2 bf16, rows of 32 bf16 = 16 words)
    const unsigned int* src = (const unsigned int*)(Kh + (size_t)bh * SS * DD);
    unsigned int* dst = (unsigned int*)sK;
    for (int w = tid; w < SKEYS * 16; w += 256) {
      int key = w >> 4;
      dst[w] = (key < SS) ? src[w] : 0u;
    }
  }
  { // cooperative stage of V^T tile
    const bf16_t* Vbh = Vt + (size_t)bh * DD * VLD;
    unsigned int* dst = (unsigned int*)sV;
    for (int w = tid; w < DD * (SKEYS / 2); w += 256) {
      int d = w / (SKEYS / 2);
      int pr = w - d * (SKEYS / 2);
      unsigned int val = 0u;
      if (2 * pr < SS) val = ((const unsigned int*)(Vbh + (size_t)d * VLD))[pr];
      dst[w] = val;
    }
  }
  __syncthreads();

  const int wave = tid >> 5, lane = tid & 31;
  const int sel = lane >> 4, ln = lane & 15, rb = sel * 8;
  const float scale = 0.17677669529663687f;  // 1/sqrt(32)
  bf16_t* myP = &sP[wave][0];
  const bf16_t* Qbh = Qh + (size_t)bh * SS * DD;
  const unsigned char* maskb = kmask + (size_t)b * SS;

  for (int qt = wave; qt < 13; qt += 8) {   // 13 query tiles of 16 (208 >= 200)
    int srow = qt * 16 + ln;
    if (srow > SS - 1) srow = SS - 1;       // clamp pad rows (results discarded)
    v16bf Aq = a_from_bf16(Qbh + (size_t)srow * DD, sel);

    float mX[8], lX[8];
    v8f acc0 = zero8(), acc1 = zero8();
#pragma unroll
    for (int r = 0; r < 8; ++r) { mX[r] = -__builtin_inff(); lX[r] = 0.0f; }

    for (int kc = 0; kc < 7; ++kc) {
      const int base = kc * 32;
      // B = K^T : lane column = key, K-dim = d (contiguous in sK rows)
      v16bf Bk0 = *(const v16bf*)(sK + (size_t)(base + ln) * DD + sel * 16);
      v16bf Bk1 = *(const v16bf*)(sK + (size_t)(base + 16 + ln) * DD + sel * 16);
      v8f s0 = WMMA_BF16(Aq, Bk0, zero8());
      v8f s1 = WMMA_BF16(Aq, Bk1, zero8());

      const int k0 = base + ln, k1 = k0 + 16;
      const bool ok0 = (k0 < SS) && (maskb[k0 < SS ? k0 : 0] != 0);
      const bool ok1 = (k1 < SS) && (maskb[k1 < SS ? k1 : 0] != 0);

      v8f p0, p1;
#pragma unroll
      for (int r = 0; r < 8; ++r) {
        float a0 = ok0 ? s0[r] * scale : -__builtin_inff();
        float a1 = ok1 ? s1[r] * scale : -__builtin_inff();
        float t = red_max16(fmaxf(a0, a1));
        float mn = fmaxf(mX[r], t);
        float msub = fmaxf(mn, -1.0e30f);       // finite exp offset: no NaN on all-masked
        float corr = __expf(mX[r] - msub);
        float e0 = __expf(a0 - msub);
        float e1 = __expf(a1 - msub);
        float rs = red_sum16(e0 + e1);
        lX[r] = lX[r] * corr + rs;
        mX[r] = mn;
        acc0[r] *= corr;
        acc1[r] *= corr;
        p0[r] = e0;
        p1[r] = e1;
      }

      // re-layout P (C-layout -> A-layout) through per-wave LDS; DS ops of a
      // wave are in-order, so no barrier needed.
#pragma unroll
      for (int r = 0; r < 8; ++r) {
        myP[(rb + r) * 32 + ln]      = (bf16_t)p0[r];
        myP[(rb + r) * 32 + 16 + ln] = (bf16_t)p1[r];
      }
      v16bf Ap = a_from_bf16(myP + (size_t)ln * 32, sel);

      // B = V : lane column = d, K-dim = keys (contiguous in sV rows)
      const bf16_t* Vb = sV + base;
      v16bf Bv0 = *(const v16bf*)(Vb + (size_t)ln * SKEYS + sel * 16);
      v16bf Bv1 = *(const v16bf*)(Vb + (size_t)(16 + ln) * SKEYS + sel * 16);
      acc0 = WMMA_BF16(Ap, Bv0, acc0);
      acc1 = WMMA_BF16(Ap, Bv1, acc1);
    }

    // epilogue: normalize, add residual (queries), store fp32
#pragma unroll
    for (int r = 0; r < 8; ++r) {
      int s = qt * 16 + rb + r;
      if (s < SS) {
        float inv = 1.0f / lX[r];
        size_t ro = ((size_t)b * SS + s) * EE + h * DD;
        outf[ro + ln]      = acc0[r] * inv + queries[ro + ln];
        outf[ro + 16 + ln] = acc1[r] * inv + queries[ro + 16 + ln];
      }
    }
  }
}

// ---------- kernel 3: fused FFN (relu(x W1^T + b1) W2^T + b2 + x) ----------
__global__ __launch_bounds__(128) void ffn_kernel(
    const float* __restrict__ outf,
    const unsigned short* __restrict__ W1_, const float* __restrict__ b1,
    const unsigned short* __restrict__ W2_, const float* __restrict__ b2,
    float* __restrict__ y) {
  __shared__ bf16_t sH[4][16 * FFD];       // per-wave hidden activations (64 KB)
  const bf16_t* W1 = (const bf16_t*)W1_;
  const bf16_t* W2 = (const bf16_t*)W2_;
  const int wave = threadIdx.x >> 5, lane = threadIdx.x & 31;
  const int sel = lane >> 4, ln = lane & 15, rb = sel * 8;
  const int m0 = blockIdx.x * 64 + wave * 16;
  const float* Arow = outf + (size_t)(m0 + ln) * EE;
  bf16_t* myH = &sH[wave][0];

  // preload the wave's 16x128 A tile (fp32 -> bf16)
  v16bf A[4];
#pragma unroll
  for (int ks = 0; ks < 4; ++ks) A[ks] = a_from_f32(Arow + ks * 32, sel);

  // stage 1: hidden = relu(x @ W1^T + b1), N=512 in chunks of 64
  for (int nc = 0; nc < 8; ++nc) {
    v8f acc[4];
#pragma unroll
    for (int t = 0; t < 4; ++t) acc[t] = zero8();
#pragma unroll
    for (int ks = 0; ks < 4; ++ks) {
#pragma unroll
      for (int t = 0; t < 4; ++t) {
        const int n = nc * 64 + t * 16 + ln;
        const bf16_t* Bp = W1 + (size_t)n * EE + ks * 32 + sel * 16;
        acc[t] = WMMA_BF16(A[ks], *(const v16bf*)Bp, acc[t]);
      }
    }
#pragma unroll
    for (int t = 0; t < 4; ++t) {
      const int n = nc * 64 + t * 16 + ln;
      const float bn = b1[n];
#pragma unroll
      for (int r = 0; r < 8; ++r) {
        float v = acc[t][r] + bn;
        v = v > 0.0f ? v : 0.0f;
        myH[(rb + r) * FFD + n] = (bf16_t)v;   // wave-local LDS, DS in-order
      }
    }
  }

  // stage 2: y = hidden @ W2^T + b2 + x, K=512
  v8f acc2[8];
#pragma unroll
  for (int t = 0; t < 8; ++t) acc2[t] = zero8();
  for (int ks = 0; ks < 16; ++ks) {
    v16bf Ah = a_from_bf16(myH + (size_t)ln * FFD + ks * 32, sel);
#pragma unroll
    for (int t = 0; t < 8; ++t) {
      const int n = t * 16 + ln;
      const bf16_t* Bp = W2 + (size_t)n * FFD + ks * 32 + sel * 16;
      acc2[t] = WMMA_BF16(Ah, *(const v16bf*)Bp, acc2[t]);
    }
  }
#pragma unroll
  for (int t = 0; t < 8; ++t) {
    const int n = t * 16 + ln;
    const float bn = b2[t * 16 + ln];
#pragma unroll
    for (int r = 0; r < 8; ++r) {
      const size_t idx = (size_t)(m0 + rb + r) * EE + n;
      y[idx] = acc2[t][r] + bn + outf[idx];
    }
  }
}

// ---------- launcher ----------
extern "C" void kernel_launch(void* const* d_in, const int* in_sizes, int n_in,
                              void* d_out, int out_size, void* d_ws, size_t ws_size,
                              hipStream_t stream) {
  const float* queries = (const float*)d_in[0];
  const float* keys    = (const float*)d_in[1];
  const unsigned char* kmask = (const unsigned char*)d_in[2];
  const float* Wq = (const float*)d_in[3];
  const float* bq = (const float*)d_in[4];
  const float* Wk = (const float*)d_in[5];
  const float* bk = (const float*)d_in[6];
  const float* Wv = (const float*)d_in[7];
  const float* bv = (const float*)d_in[8];
  const float* W1 = (const float*)d_in[9];
  const float* b1 = (const float*)d_in[10];
  const float* W2 = (const float*)d_in[11];
  const float* b2 = (const float*)d_in[12];
  float* y = (float*)d_out;

  char* ws = (char*)d_ws;
  size_t off = 0;
  auto carve = [&](size_t bytes) -> void* {
    void* p = (void*)(ws + off);
    off += (bytes + 255) & ~(size_t)255;
    return p;
  };
  unsigned short* Wq_b = (unsigned short*)carve((size_t)EE * EE * 2);
  unsigned short* Wk_b = (unsigned short*)carve((size_t)EE * EE * 2);
  unsigned short* Wv_b = (unsigned short*)carve((size_t)EE * EE * 2);
  unsigned short* W1_b = (unsigned short*)carve((size_t)FFD * EE * 2);
  unsigned short* W2_b = (unsigned short*)carve((size_t)EE * FFD * 2);
  unsigned short* Qh   = (unsigned short*)carve((size_t)BHNUM * SS * DD * 2);
  unsigned short* Kh   = (unsigned short*)carve((size_t)BHNUM * SS * DD * 2);
  unsigned short* Vt   = (unsigned short*)carve((size_t)BHNUM * DD * VLD * 2);
  float* outf          = (float*)carve((size_t)MROWS * EE * 4);

  cvt_f32_bf16<<<(EE * EE + 255) / 256, 256, 0, stream>>>(Wq, Wq_b, EE * EE);
  cvt_f32_bf16<<<(EE * EE + 255) / 256, 256, 0, stream>>>(Wk, Wk_b, EE * EE);
  cvt_f32_bf16<<<(EE * EE + 255) / 256, 256, 0, stream>>>(Wv, Wv_b, EE * EE);
  cvt_f32_bf16<<<(FFD * EE + 255) / 256, 256, 0, stream>>>(W1, W1_b, FFD * EE);
  cvt_f32_bf16<<<(EE * FFD + 255) / 256, 256, 0, stream>>>(W2, W2_b, EE * FFD);

  qkv_proj_kernel<<<dim3(MROWS / 64, 3), 128, 0, stream>>>(
      queries, keys, Wq_b, Wk_b, Wv_b, bq, bk, bv, Qh, Kh, Vt);
  attn_kernel<<<BHNUM, 256, 0, stream>>>(Qh, Kh, Vt, kmask, queries, outf);
  ffn_kernel<<<MROWS / 64, 128, 0, stream>>>(outf, W1_b, b1, W2_b, b2, y);

  (void)in_sizes; (void)n_in; (void)out_size; (void)ws_size;
}
